// MMDDistance_29729763623082
// MI455X (gfx1250) — matrix operand order repaired and checked by reading, not compiled, not run on hardware
//
#include <hip/hip_runtime.h>

// MMD distance for MI455X (gfx1250), wave32.
//
// support_xf: [4, 5, 196, 64] f32   query_xf: [4, 75, 196, 64] f32
// out:        [4, 75, 5] f32
//
// Gram matrices computed with V_WMMA_F32_16X16X32_BF16 using a 2-term
// bf16 split (x = hi + lo): x.y ~= hi.hi + hi.lo + lo.hi, accumulated in
// fp32 (error ~2^-16 relative -> negligible through exp(-alpha d^2)).
// 16-bit WMMA is tracked as a TRANS op on CDNA5, so it co-executes with
// the heavy VALU exp epilogue. The 5-alpha Gaussian mixture collapses to
// one v_exp plus a squaring chain (alphas are powers of two).

typedef __bf16 bf16_t;
typedef __attribute__((ext_vector_type(8)))  __bf16 v8bf;
typedef __attribute__((ext_vector_type(16))) __bf16 v16bf;
typedef __attribute__((ext_vector_type(8)))  float  v8f;

#define NF     196          // features per set
#define NC     64           // channels (K)
#define NPAD   208          // 13 * 16 (tile-padded rows)
#define LDB    72           // bf16 row stride (144B = 36 dwords): all 16
                            // bank-quads distinct per b128 read phase
#define NT     13           // 16x16 tiles per dimension
#define NTILES (NT * NT)
#define TPB    256          // 8 waves

// ---- bf16 helpers -------------------------------------------------------

__device__ __forceinline__ unsigned short f2bf(float f) {
    unsigned u = __float_as_uint(f);
    u += 0x7fffu + ((u >> 16) & 1u);      // round-to-nearest-even
    return (unsigned short)(u >> 16);
}
__device__ __forceinline__ float bf2f(unsigned short h) {
    return __uint_as_float((unsigned)h << 16);
}

// sum_{k=-3..1} exp(-2^k * d2) ; caller divides by 5.
__device__ __forceinline__ float multi_gauss_sum(float d2) {
    float e   = __expf(-0.125f * d2);
    float p2  = e * e;
    float p4  = p2 * p2;
    float p8  = p4 * p4;
    float p16 = p8 * p8;
    return e + p2 + p4 + p8 + p16;
}

// 196x64 f32 global -> hi/lo bf16 planes, 208x(LDB) padded, rows>=196 zero.
__device__ __forceinline__ void load_mat_split(const float* __restrict__ g,
                                               unsigned short* hi,
                                               unsigned short* lo) {
    for (int idx = threadIdx.x; idx < NPAD * (NC / 4); idx += TPB) {
        int row = idx >> 4;
        int c4  = (idx & 15) << 2;
        float4 v = make_float4(0.f, 0.f, 0.f, 0.f);
        if (row < NF) v = *(const float4*)(g + row * NC + c4);
        ushort4 h, l;
        h.x = f2bf(v.x); l.x = f2bf(v.x - bf2f(h.x));
        h.y = f2bf(v.y); l.y = f2bf(v.y - bf2f(h.y));
        h.z = f2bf(v.z); l.z = f2bf(v.z - bf2f(h.z));
        h.w = f2bf(v.w); l.w = f2bf(v.w - bf2f(h.w));
        *(ushort4*)(hi + row * LDB + c4) = h;   // 8B aligned
        *(ushort4*)(lo + row * LDB + c4) = l;
    }
}

// Row norms from the same hi+lo reconstruction the matmul sees.
__device__ __forceinline__ void compute_norms_split(const unsigned short* hi,
                                                    const unsigned short* lo,
                                                    float* sn) {
    for (int row = threadIdx.x; row < NPAD; row += TPB) {
        float acc = 0.f;
        #pragma unroll 8
        for (int c = 0; c < NC; ++c) {
            float x = bf2f(hi[row * LDB + c]) + bf2f(lo[row * LDB + c]);
            acc += x * x;
        }
        sn[row] = acc;
    }
}

// Two 16B LDS chunks (gap elements apart) -> one 16-element bf16 operand.
__device__ __forceinline__ v16bf ld2(const bf16_t* p, int gap) {
    v8bf x = *(const v8bf*)(p);
    v8bf y = *(const v8bf*)(p + gap);
    return __builtin_shufflevector(x, y, 0, 1, 2, 3, 4, 5, 6, 7,
                                         8, 9, 10, 11, 12, 13, 14, 15);
}

// Per-wave pass over the 13x13 tile grid of the Gram matrix A.B^T.
template <bool EXCL_DIAG>
__device__ __forceinline__ float tile_pass(const unsigned short* aHi,
                                           const unsigned short* aLo,
                                           const unsigned short* bHi,
                                           const unsigned short* bLo,
                                           const float* sAn, const float* sBn) {
    const int lane = threadIdx.x & 31;
    const int wave = threadIdx.x >> 5;
    const int hf   = lane >> 4;      // lane half selects K sub-blocks
    const int lr   = lane & 15;      // row (A) / col (B) within tile

    float acc = 0.f;
    for (int t = wave; t < NTILES; t += (TPB >> 5)) {
        const int ti = t / NT;
        const int tj = t % NT;
        const bf16_t* Ah = (const bf16_t*)(aHi + (ti * 16 + lr) * LDB);
        const bf16_t* Al = (const bf16_t*)(aLo + (ti * 16 + lr) * LDB);
        const bf16_t* Bh = (const bf16_t*)(bHi + (tj * 16 + lr) * LDB);
        const bf16_t* Bl = (const bf16_t*)(bLo + (tj * 16 + lr) * LDB);

        v8f c = {0.f, 0.f, 0.f, 0.f, 0.f, 0.f, 0.f, 0.f};
        #pragma unroll
        for (int kb = 0; kb < NC; kb += 32) {
            // A 16x32 bf16: VGPR0-3 = K[8*hf..+7], VGPR4-7 = K[16+8*hf..+7]
            v16bf ah = ld2(Ah + kb + 8 * hf, 16);
            v16bf al = ld2(Al + kb + 8 * hf, 16);
            // B 32x16 bf16: VGPR0-3 = K[8*hf..+7], VGPR4-7 = K[16+8*hf..+7]
            v16bf bh = ld2(Bh + kb + 8 * hf, 16);
            v16bf bl = ld2(Bl + kb + 8 * hf, 16);
            // split product: hi.hi + hi.lo + lo.hi (fp32 accumulate)
            c = __builtin_amdgcn_wmma_f32_16x16x32_bf16(
                    false, ah, false, bh, (short)0, c, false, false);
            c = __builtin_amdgcn_wmma_f32_16x16x32_bf16(
                    false, ah, false, bl, (short)0, c, false, false);
            c = __builtin_amdgcn_wmma_f32_16x16x32_bf16(
                    false, al, false, bh, (short)0, c, false, false);
        }

        // C/D layout: VGPR v, lane L: N = L%16, M = v + 8*(L/16)
        const int j   = tj * 16 + lr;
        const float bn = sBn[j];
        #pragma unroll
        for (int v = 0; v < 8; ++v) {
            const int i  = ti * 16 + v + 8 * hf;
            float d2 = fmaxf(sAn[i] + bn - 2.0f * c[v], 0.f);
            float ks = multi_gauss_sum(d2);
            bool ok = (i < NF) && (j < NF);
            if (EXCL_DIAG) ok = ok && (i != j);
            acc += ok ? ks : 0.f;
        }
    }
    return acc;
}

// Deterministic block tree-reduction (no float atomics -> replay-stable).
__device__ __forceinline__ float block_reduce(float v, float* red) {
    const int tid = threadIdx.x;
    red[tid] = v;
    __syncthreads();
    for (int s = TPB >> 1; s > 0; s >>= 1) {
        if (tid < s) red[tid] += red[tid + s];
        __syncthreads();
    }
    return red[0];
}

#define PLANE (NPAD * LDB)   // elements per bf16 plane

// --- Kernel 1: off-diagonal mean of multi-gauss Gram for one set ---
__global__ void __launch_bounds__(TPB)
mmd_within_kernel(const float* __restrict__ x, float* __restrict__ out_mmd) {
    extern __shared__ unsigned char smem_raw[];
    unsigned short* sHi = (unsigned short*)smem_raw;        // PLANE
    unsigned short* sLo = sHi + PLANE;                      // PLANE
    float* sN  = (float*)(sLo + PLANE);                     // NPAD
    float* red = sN + NPAD;                                 // TPB

    load_mat_split(x + (size_t)blockIdx.x * NF * NC, sHi, sLo);
    __syncthreads();
    compute_norms_split(sHi, sLo, sN);
    __syncthreads();

    float acc = tile_pass<true>(sHi, sLo, sHi, sLo, sN, sN);
    float tot = block_reduce(acc, red);
    if (threadIdx.x == 0)
        out_mmd[blockIdx.x] = tot * 0.2f / (float)(NF * (NF - 1));
}

// --- Kernel 2: cross term + final combine ---
// grid.x = 4*75*5 ; block g = (b*75 + q)*5 + s == flat out index
__global__ void __launch_bounds__(TPB)
mmd_cross_kernel(const float* __restrict__ support,
                 const float* __restrict__ query,
                 const float* __restrict__ mmd_s,   // [4*5]
                 const float* __restrict__ mmd_q,   // [4*75]
                 float* __restrict__ out) {
    extern __shared__ unsigned char smem_raw[];
    unsigned short* sHiS = (unsigned short*)smem_raw;       // PLANE
    unsigned short* sLoS = sHiS + PLANE;                    // PLANE
    unsigned short* sHiQ = sLoS + PLANE;                    // PLANE
    unsigned short* sLoQ = sHiQ + PLANE;                    // PLANE
    float* sSn = (float*)(sLoQ + PLANE);                    // NPAD
    float* sQn = sSn + NPAD;                                // NPAD
    float* red = sQn + NPAD;                                // TPB

    const int g = blockIdx.x;
    const int s = g % 5;
    const int q = (g / 5) % 75;
    const int b = g / (5 * 75);

    load_mat_split(support + (size_t)(b * 5 + s) * NF * NC, sHiS, sLoS);
    load_mat_split(query   + (size_t)(b * 75 + q) * NF * NC, sHiQ, sLoQ);
    __syncthreads();
    compute_norms_split(sHiS, sLoS, sSn);
    compute_norms_split(sHiQ, sLoQ, sQn);
    __syncthreads();

    float acc = tile_pass<false>(sHiS, sLoS, sHiQ, sLoQ, sSn, sQn);
    float tot = block_reduce(acc, red);
    if (threadIdx.x == 0) {
        float mmd_sq = tot * 0.2f / (float)(NF * NF);
        out[g] = (mmd_s[b * 5 + s] + mmd_q[b * 75 + q] - 2.0f * mmd_sq) * 12.5f;
    }
}

extern "C" void kernel_launch(void* const* d_in, const int* in_sizes, int n_in,
                              void* d_out, int out_size, void* d_ws, size_t ws_size,
                              hipStream_t stream) {
    const float* support = (const float*)d_in[0];  // [4,5,196,64]
    const float* query   = (const float*)d_in[1];  // [4,75,196,64]
    float* out = (float*)d_out;                    // [4,75,5]

    float* ws_ms = (float*)d_ws;   // 20 floats
    float* ws_mq = ws_ms + 20;     // 300 floats

    const size_t sh_within = (size_t)(2 * PLANE) * sizeof(unsigned short)
                           + (size_t)(NPAD + TPB) * sizeof(float);
    const size_t sh_cross  = (size_t)(4 * PLANE) * sizeof(unsigned short)
                           + (size_t)(2 * NPAD + TPB) * sizeof(float);

    // within-set MMD terms
    mmd_within_kernel<<<20,  TPB, sh_within, stream>>>(support, ws_ms);
    mmd_within_kernel<<<300, TPB, sh_within, stream>>>(query,   ws_mq);
    // cross term + combine (reads ws written above; same-stream ordering)
    mmd_cross_kernel<<<1500, TPB, sh_cross, stream>>>(support, query,
                                                      ws_ms, ws_mq, out);
}
